// Re4_34110630265582
// MI455X (gfx1250) — compile-verified, matrix operand order checked by reading, not run on hardware
//
#include <hip/hip_runtime.h>
#include <math.h>

#define BB 64
#define SS 256
#define DD 64
#define HH 64
#define NIk 8
#define H4 256
#define PADV (-4294967295.0f)

typedef __attribute__((ext_vector_type(16))) _Float16 v16h;
typedef __attribute__((ext_vector_type(8)))  float    v8f;

// ---------- WMMA fragment helpers (16x16x32 f16 -> f32) ----------
// 16-bit A 16x32 layout (ISA 7.12.2): lanes 0-15 = M rows, half-wave splits K.
__device__ __forceinline__ int kslot(int v, int h) {
  int base = (v < 4) ? (2 * v) : (16 + 2 * (v - 4));
  return base + 8 * h;
}

// A fragment: element (m,k) at p[m*rs + k*cs]
__device__ __forceinline__ v16h frag_a(const _Float16* p, int rs, int cs) {
  int l = threadIdx.x & 31;
  int m = l & 15, h = l >> 4;
  v16h f;
#pragma unroll
  for (int v = 0; v < 8; ++v) {
    int k = kslot(v, h);
    f[2 * v]     = p[m * rs + k * cs];
    f[2 * v + 1] = p[m * rs + (k + 1) * cs];
  }
  return f;
}

// B fragment (32x16, lane holds column n): element (k,n) at p[k*rs + n*cs]
// All call sites use rs==1 (K-innermost storage) so each lane issues two
// contiguous 16-byte runs -> global_load_b128.
__device__ __forceinline__ v16h frag_b(const _Float16* p, int rs, int cs) {
  int l = threadIdx.x & 31;
  int n = l & 15, h = l >> 4;
  v16h f;
#pragma unroll
  for (int v = 0; v < 8; ++v) {
    int k = kslot(v, h);
    f[2 * v]     = p[k * rs + n * cs];
    f[2 * v + 1] = p[(k + 1) * rs + n * cs];
  }
  return f;
}

__device__ __forceinline__ v8f wmma32(v16h a, v16h b, v8f c) {
  return __builtin_amdgcn_wmma_f32_16x16x32_f16(false, a, false, b, (short)0, c,
                                                false, false);
}

// ---------- block reductions ----------
template <int N>
__device__ __forceinline__ float blk_sum(float v, float* red) {
  int t = threadIdx.x;
  red[t] = v; __syncthreads();
#pragma unroll
  for (int o = N >> 1; o > 0; o >>= 1) {
    if (t < o) red[t] += red[t + o];
    __syncthreads();
  }
  float r = red[0]; __syncthreads();
  return r;
}
template <int N>
__device__ __forceinline__ float blk_max(float v, float* red) {
  int t = threadIdx.x;
  red[t] = v; __syncthreads();
#pragma unroll
  for (int o = N >> 1; o > 0; o >>= 1) {
    if (t < o) red[t] = fmaxf(red[t], red[t + o]);
    __syncthreads();
  }
  float r = red[0]; __syncthreads();
  return r;
}

// ---------- kernels ----------
// Transposing f32->f16 weight convert: dst[n*64 + k] = src[k*N + n], k<64.
__global__ void k_cvt_t(const float* src, _Float16* dst, int N) {
  int j = blockIdx.x * blockDim.x + threadIdx.x;
  if (j >= N * 64) return;
  int n = j >> 6, k = j & 63;
  dst[j] = (_Float16)src[(size_t)k * N + n];
}

// item_his = emb[mid_his]*mask ; x = item_his + pos_emb (f16) ; iln = l2norm(item_his)
__global__ void k_gather(const int* mid_his, const float* mask, const float* emb,
                         const float* pos_emb, float* item_his, _Float16* xh,
                         float* iln) {
  __shared__ float red[64];
  int bs = blockIdx.x, d = threadIdx.x;
  int mid = mid_his[bs];
  float msk = mask[bs];
  float e = emb[(size_t)mid * DD + d] * msk;
  item_his[(size_t)bs * DD + d] = e;
  xh[(size_t)bs * DD + d] = (_Float16)(e + pos_emb[(bs % SS) * DD + d]);
  float ss = blk_sum<64>(e * e, red);
  iln[(size_t)bs * DD + d] = e * rsqrtf(fmaxf(ss, 1e-12f));
}

// hid = tanh(x @ W1 + b1) : (16384x64)@(64x256), WMMA (W1 stored transposed)
__global__ void k_hid(const _Float16* X, const _Float16* w1t, const float* b1,
                      _Float16* hid) {
  int gw = (blockIdx.x * blockDim.x + threadIdx.x) >> 5;
  int nt = gw % (H4 / 16), mt = gw / (H4 / 16);
  int m0 = mt * 16, n0 = nt * 16;
  v8f c = {};
#pragma unroll
  for (int k0 = 0; k0 < DD; k0 += 32) {
    v16h a = frag_a(X + (size_t)m0 * DD + k0, DD, 1);
    v16h b = frag_b(w1t + (size_t)n0 * DD + k0, 1, DD);  // w1t[n][k]
    c = wmma32(a, b, c);
  }
  int l = threadIdx.x & 31;
  int col = n0 + (l & 15), r0 = m0 + 8 * (l >> 4);
  float bias = b1[col];
#pragma unroll
  for (int i = 0; i < 8; ++i)
    hid[(size_t)(r0 + i) * H4 + col] = (_Float16)tanhf(c[i] + bias);
}

// logits[b,k,s] = hid[b,s,:] . W2[:,k] + b2[k], masked -> PAD
__global__ void k_logits(const _Float16* hid, const float* W2, const float* b2,
                         const float* mask, float* logits) {
  int idx = blockIdx.x * blockDim.x + threadIdx.x;
  if (idx >= BB * NIk * SS) return;
  int s = idx % SS, k = (idx / SS) % NIk, b = idx / (SS * NIk);
  const _Float16* h = hid + (size_t)(b * SS + s) * H4;
  float acc = b2[k];
  for (int j = 0; j < H4; ++j) acc += (float)h[j] * W2[j * NIk + k];
  logits[idx] = (mask[b * SS + s] != 0.f) ? acc : PADV;
}

__global__ void k_softmax(const float* logits, float* pw, float* att) {
  __shared__ float red[256];
  int bk = blockIdx.x, s = threadIdx.x;
  float v = logits[(size_t)bk * SS + s];
  float mx = blk_max<256>(v, red);
  float e = expf(v - mx);
  float sum = blk_sum<256>(e, red);
  pw[(size_t)bk * SS + s]  = e / sum;
  att[(size_t)bk * SS + s] = v - mx - logf(sum);
}

// wi = pw @ item_his ; wn = l2norm(wi)
__global__ void k_wi(const float* pw, const float* item_his, float* wi,
                     _Float16* wih, float* wn) {
  __shared__ float red[64];
  int bk = blockIdx.x, d = threadIdx.x, b = bk / NIk;
  float acc = 0.f;
  for (int s = 0; s < SS; ++s)
    acc += pw[(size_t)bk * SS + s] * item_his[(size_t)(b * SS + s) * DD + d];
  wi[(size_t)bk * DD + d]  = acc;
  wih[(size_t)bk * DD + d] = (_Float16)acc;
  float ss = blk_sum<64>(acc * acc, red);
  wn[(size_t)bk * DD + d] = acc * rsqrtf(fmaxf(ss, 1e-12f));
}

// re_att softmax + partial sums for loss_attend
__global__ void k_attend(const float* wi, const float* item_his, const float* mask,
                         const float* att, float* pnum, float* pden) {
  __shared__ float red[256];
  int bk = blockIdx.x, s = threadIdx.x, b = bk / NIk;
  float dot = 0.f;
  for (int d = 0; d < DD; ++d)
    dot += wi[(size_t)bk * DD + d] * item_his[(size_t)(b * SS + s) * DD + d];
  float v = (mask[b * SS + s] != 0.f) ? dot : PADV;
  float mx = blk_max<256>(v, red);
  float e = expf(v - mx);
  float sum = blk_sum<256>(e, red);
  float re = e / sum;
  float n = blk_sum<256>(re * att[(size_t)bk * SS + s], red);
  float dn = blk_sum<256>(re, red);
  if (s == 0) { pnum[bk] = n; pden[bk] = dn; }
}

// contrastive loss partials + pwi gate
__global__ void k_contrast(const float* wn, const float* iln, const float* pw,
                           const float* mask, const int* mid_his, float* pwi,
                           float* pcon) {
  __shared__ float red[256];
  int bk = blockIdx.x, s = threadIdx.x;
  int b = bk / NIk, k = bk % NIk;
  float msk = mask[b * SS + s];
  float mlen = blk_sum<256>(msk, red);
  int bp = (b + BB - 1) % BB;  // roll(+1) source
  float cosv = 0.f, cos2 = 0.f;
  for (int d = 0; d < DD; ++d) {
    float w = wn[(size_t)bk * DD + d];
    cosv += w * iln[(size_t)(b * SS + s) * DD + d];
    cos2 += w * iln[(size_t)(bp * SS + s) * DD + d];
  }
  float mc = (msk != 0.f) ? cosv : PADV;
  float gate = 1.f / mlen;
  float pv = (pw[(size_t)bk * SS + s] > gate) ? 1.f : 0.f;
  pwi[(size_t)bk * SS + s] = pv;
  if (mid_his[bp * SS + s] == 0) cos2 = PADV;
  float neglocal = expf(mc) + expf(cos2);     // expf(PAD) == 0
  if (s < NIk && s != k) {                    // in2in off-diagonal terms
    float d2 = 0.f;
    for (int d = 0; d < DD; ++d)
      d2 += wn[(size_t)bk * DD + d] * wn[(size_t)(b * NIk + s) * DD + d];
    neglocal += expf(d2);
  }
  float neg = blk_sum<256>(neglocal, red);
  float contrib = (pv != 0.f && msk != 0.f) ? -(mc - logf(neg)) : 0.f;
  float csum = blk_sum<256>(contrib, red);
  if (s == 0) pcon[bk] = csum;
}

// recons = wi @ Wr + br : (512x64)@(64x16384), WMMA (Wr stored transposed).
// Writes row-major reconsh[(bk)*16384 + s*64+d] and transposed reconsT[bk][d][s].
__global__ void k_recons(const _Float16* wih, const _Float16* wrt, const float* br,
                         _Float16* reconsh, _Float16* reconsT) {
  int gw = (blockIdx.x * blockDim.x + threadIdx.x) >> 5;
  const int NT = (HH * SS) / 16;  // 1024
  int nt = gw % NT, mt = gw / NT;
  int m0 = mt * 16, n0 = nt * 16;
  v8f c = {};
#pragma unroll
  for (int k0 = 0; k0 < DD; k0 += 32) {
    v16h a = frag_a(wih + (size_t)m0 * DD + k0, DD, 1);
    v16h b = frag_b(wrt + (size_t)n0 * DD + k0, 1, DD);  // wrt[e][d]
    c = wmma32(a, b, c);
  }
  int l = threadIdx.x & 31;
  int col = n0 + (l & 15), r0 = m0 + 8 * (l >> 4);
  int s = col >> 6, d = col & 63;
  float bias = br[col];
#pragma unroll
  for (int i = 0; i < 8; ++i) {
    _Float16 v = (_Float16)(c[i] + bias);
    reconsh[(size_t)(r0 + i) * (HH * SS) + col] = v;
    reconsT[((size_t)(r0 + i) * DD + d) * SS + s] = v;
  }
}

// rh = tanh(recons @ Wh + bh): (131072x64)@(64x64), WMMA (Wh stored transposed)
__global__ void k_rh(const _Float16* reconsh, const _Float16* wht, const float* bh,
                     _Float16* rhh) {
  int gw = (blockIdx.x * blockDim.x + threadIdx.x) >> 5;
  int nt = gw % (HH / 16), mt = gw / (HH / 16);
  int m0 = mt * 16, n0 = nt * 16;
  v8f c = {};
#pragma unroll
  for (int k0 = 0; k0 < DD; k0 += 32) {
    v16h a = frag_a(reconsh + (size_t)m0 * DD + k0, DD, 1);
    v16h b = frag_b(wht + (size_t)n0 * DD + k0, 1, DD);  // wht[h_out][d]
    c = wmma32(a, b, c);
  }
  int l = threadIdx.x & 31;
  int col = n0 + (l & 15), r0 = m0 + 8 * (l >> 4);
  float bias = bh[col];
#pragma unroll
  for (int i = 0; i < 8; ++i)
    rhh[(size_t)(r0 + i) * HH + col] = (_Float16)tanhf(c[i] + bias);
}

// flash pass 1: per (b,k,t-tile) row max & exp-sum of
// score[t,s] = Ww[:,t].rh[b,k,s,:] + bw[t]   (wwt = Ww^T, t-major)
__global__ void k_sstats(const _Float16* rhh, const _Float16* wwt, const float* bw,
                         const int* mid_his, float* stats) {
  int gw = (blockIdx.x * blockDim.x + threadIdx.x) >> 5;
  int tt = gw & 15, bk = gw >> 4, b = bk / NIk;
  int t0 = tt * 16;
  const _Float16* rb = rhh + (size_t)bk * SS * HH;
  int l = threadIdx.x & 31, g = l >> 4, nl = l & 15;
  // loop-invariant A fragments (Ww^T rows t0..t0+15)
  v16h aw0 = frag_a(wwt + (size_t)t0 * DD + 0,  DD, 1);
  v16h aw1 = frag_a(wwt + (size_t)t0 * DD + 32, DD, 1);
  float m[8], ls[8];
#pragma unroll
  for (int i = 0; i < 8; ++i) { m[i] = -INFINITY; ls[i] = 0.f; }
  for (int s0 = 0; s0 < SS; s0 += 16) {
    v8f c = {};
    c = wmma32(aw0, frag_b(rb + (size_t)s0 * HH + 0,  1, HH), c);
    c = wmma32(aw1, frag_b(rb + (size_t)s0 * HH + 32, 1, HH), c);
    bool dead = (mid_his[b * SS + s0 + nl] == 0);
#pragma unroll
    for (int i = 0; i < 8; ++i) {
      float v = c[i] + bw[t0 + i + 8 * g];
      if (dead) v = PADV;
      float tm = v;
      for (int o = 8; o >= 1; o >>= 1) tm = fmaxf(tm, __shfl_xor(tm, o, 16));
      float nmax = fmaxf(m[i], tm);
      float e = expf(v - nmax);
      float ts = e;
      for (int o = 8; o >= 1; o >>= 1) ts += __shfl_xor(ts, o, 16);
      ls[i] = ls[i] * expf(m[i] - nmax) + ts;
      m[i] = nmax;
    }
  }
  if (nl == 0) {
#pragma unroll
    for (int i = 0; i < 8; ++i) {
      int t = t0 + i + 8 * g;
      stats[((size_t)bk * SS + t) * 2 + 0] = m[i];
      stats[((size_t)bk * SS + t) * 2 + 1] = ls[i];
    }
  }
}

// flash pass 2: recompute scores, p=softmax, out = p @ recons ; fused construct-loss
__global__ void k_sout(const _Float16* rhh, const _Float16* wwt, const float* bw,
                       const int* mid_his, const float* stats,
                       const _Float16* reconsT, const float* item_his,
                       const float* pwi, const float* mask, float* pcons) {
  __shared__ _Float16 plds[8][16][32];
  int gw = (blockIdx.x * blockDim.x + threadIdx.x) >> 5;
  int wid = threadIdx.x >> 5;
  int tt = gw & 15, bk = gw >> 4, b = bk / NIk;
  int t0 = tt * 16;
  const _Float16* rb = rhh + (size_t)bk * SS * HH;
  const _Float16* rc = reconsT + (size_t)bk * DD * SS;  // [d][s]
  int l = threadIdx.x & 31, g = l >> 4, nl = l & 15;
  v16h aw0 = frag_a(wwt + (size_t)t0 * DD + 0,  DD, 1);
  v16h aw1 = frag_a(wwt + (size_t)t0 * DD + 32, DD, 1);
  float m[8], inv[8], bwr[8];
#pragma unroll
  for (int i = 0; i < 8; ++i) {
    int t = t0 + i + 8 * g;
    m[i]   = stats[((size_t)bk * SS + t) * 2 + 0];
    inv[i] = 1.f / stats[((size_t)bk * SS + t) * 2 + 1];
    bwr[i] = bw[t];
  }
  v8f o[4];
#pragma unroll
  for (int n = 0; n < 4; ++n)
#pragma unroll
    for (int i = 0; i < 8; ++i) o[n][i] = 0.f;

  for (int s0 = 0; s0 < SS; s0 += 32) {
    if (s0 + 32 < SS) {  // prefetch next chunk (global_prefetch_b8)
      __builtin_prefetch(rb + (size_t)(s0 + 32) * HH, 0, 3);
      __builtin_prefetch(rc + (size_t)(nl * 16) * SS + s0 + 32, 0, 3);
    }
#pragma unroll
    for (int half = 0; half < 2; ++half) {
      int sc = s0 + half * 16;
      v8f c = {};
      c = wmma32(aw0, frag_b(rb + (size_t)sc * HH + 0,  1, HH), c);
      c = wmma32(aw1, frag_b(rb + (size_t)sc * HH + 32, 1, HH), c);
      bool dead = (mid_his[b * SS + sc + nl] == 0);
#pragma unroll
      for (int i = 0; i < 8; ++i) {
        float v = c[i] + bwr[i];
        if (dead) v = PADV;
        plds[wid][i + 8 * g][half * 16 + nl] = (_Float16)(expf(v - m[i]) * inv[i]);
      }
    }
    __syncthreads();
    v16h a = frag_a(&plds[wid][0][0], 32, 1);  // p chunk 16x32 as A
#pragma unroll
    for (int n = 0; n < 4; ++n) {
      // B[k=s][n=d] from reconsT: contiguous over s per lane
      v16h bf = frag_b(rc + (size_t)(n * 16) * SS + s0, 1, SS);
      o[n] = wmma32(a, bf, o[n]);
    }
    __syncthreads();
  }
  // construct loss: sum (out - item_his[b,t,:])^2 gated by pwi & mask
  float acc = 0.f;
#pragma unroll
  for (int n = 0; n < 4; ++n)
#pragma unroll
    for (int i = 0; i < 8; ++i) {
      int t = t0 + i + 8 * g;
      int d = n * 16 + nl;
      float gt = pwi[(size_t)bk * SS + t] * ((mask[b * SS + t] != 0.f) ? 1.f : 0.f);
      float diff = o[n][i] - item_his[(size_t)(b * SS + t) * DD + d];
      acc += gt * diff * diff;
    }
  for (int oo = 16; oo >= 1; oo >>= 1) acc += __shfl_xor(acc, oo, 32);
  if (l == 0) pcons[gw] = acc;
}

__global__ void k_readout(const float* wi, const float* emb, const int* mid_batch,
                          float* out) {
  __shared__ float dots[NIk];
  __shared__ int best;
  int b = blockIdx.x, t = threadIdx.x;
  if (t < NIk) {
    const float* ie = emb + (size_t)mid_batch[b] * DD;
    float a = 0.f;
    for (int d = 0; d < DD; ++d) a += wi[((size_t)b * NIk + t) * DD + d] * ie[d];
    dots[t] = a;
  }
  __syncthreads();
  if (t == 0) {
    float mx = dots[0]; int bi = 0;
    for (int k = 1; k < NIk; ++k)
      if (dots[k] > mx) { mx = dots[k]; bi = k; }
    best = bi;
  }
  __syncthreads();
  out[(size_t)b * DD + t] = wi[((size_t)b * NIk + best) * DD + t];
}

__global__ void k_finalize(const float* pn, const float* pd, const float* pc,
                           const float* pk, float* out) {
  __shared__ float red[256];
  int t = threadIdx.x;
  float n = 0.f, d = 0.f, c = 0.f, kk = 0.f;
  for (int i = t; i < BB * NIk; i += 256) { n += pn[i]; d += pd[i]; c += pc[i]; }
  for (int i = t; i < BB * NIk * 16; i += 256) kk += pk[i];
  n  = blk_sum<256>(n, red);
  d  = blk_sum<256>(d, red);
  c  = blk_sum<256>(c, red);
  kk = blk_sum<256>(kk, red);
  if (t == 0) {
    out[BB * DD + 0] = -n / d;
    out[BB * DD + 1] = c / (float)(BB * NIk * SS);
    out[BB * DD + 2] = kk / (float)((size_t)BB * NIk * SS * DD);
  }
}

// ---------- launcher ----------
extern "C" void kernel_launch(void* const* d_in, const int* in_sizes, int n_in,
                              void* d_out, int out_size, void* d_ws, size_t ws_size,
                              hipStream_t stream) {
  const int*   mid_his   = (const int*)d_in[0];
  const int*   mid_batch = (const int*)d_in[1];
  const float* mask      = (const float*)d_in[2];
  const float* emb       = (const float*)d_in[3];
  const float* pos_emb   = (const float*)d_in[4];
  const float* W1 = (const float*)d_in[5];
  const float* b1 = (const float*)d_in[6];
  const float* W2 = (const float*)d_in[7];
  const float* b2 = (const float*)d_in[8];
  const float* Wr = (const float*)d_in[9];
  const float* br = (const float*)d_in[10];
  const float* Wh = (const float*)d_in[11];
  const float* bh = (const float*)d_in[12];
  const float* Ww = (const float*)d_in[13];
  const float* bw = (const float*)d_in[14];
  float* out = (float*)d_out;

  char* ws = (char*)d_ws;
  size_t off = 0;
  auto alloc = [&](size_t bytes) -> void* {
    off = (off + 255) & ~(size_t)255;
    void* p = ws + off;
    off += bytes;
    return p;
  };

  float*    item_his = (float*)   alloc((size_t)BB * SS * DD * 4);
  float*    iln      = (float*)   alloc((size_t)BB * SS * DD * 4);
  _Float16* xh       = (_Float16*)alloc((size_t)BB * SS * DD * 2);
  _Float16* w1t      = (_Float16*)alloc((size_t)H4 * DD * 2);        // W1^T (256x64)
  _Float16* wrt      = (_Float16*)alloc((size_t)HH * SS * DD * 2);   // Wr^T (16384x64)
  _Float16* wht      = (_Float16*)alloc((size_t)HH * DD * 2);        // Wh^T (64x64)
  _Float16* wwt      = (_Float16*)alloc((size_t)SS * DD * 2);        // Ww^T (256x64)
  _Float16* hid      = (_Float16*)alloc((size_t)BB * SS * H4 * 2);
  float*    logits   = (float*)   alloc((size_t)BB * NIk * SS * 4);
  float*    pw       = (float*)   alloc((size_t)BB * NIk * SS * 4);
  float*    att      = (float*)   alloc((size_t)BB * NIk * SS * 4);
  float*    pwi      = (float*)   alloc((size_t)BB * NIk * SS * 4);
  float*    wi       = (float*)   alloc((size_t)BB * NIk * DD * 4);
  _Float16* wih      = (_Float16*)alloc((size_t)BB * NIk * DD * 2);
  float*    wn       = (float*)   alloc((size_t)BB * NIk * DD * 4);
  _Float16* reconsh  = (_Float16*)alloc((size_t)BB * NIk * SS * DD * 2);
  _Float16* reconsT  = (_Float16*)alloc((size_t)BB * NIk * SS * DD * 2);
  _Float16* rhh      = (_Float16*)alloc((size_t)BB * NIk * SS * DD * 2);
  float*    stats    = (float*)   alloc((size_t)BB * NIk * SS * 2 * 4);
  float*    pnum     = (float*)   alloc((size_t)BB * NIk * 4);
  float*    pden     = (float*)   alloc((size_t)BB * NIk * 4);
  float*    pcon     = (float*)   alloc((size_t)BB * NIk * 4);
  float*    pcons    = (float*)   alloc((size_t)BB * NIk * 16 * 4);

  // transposed f16 weight copies (K-innermost for b128 fragment loads)
  k_cvt_t<<<(H4 * 64 + 255) / 256, 256, 0, stream>>>(W1, w1t, H4);
  k_cvt_t<<<(HH * SS * 64 + 255) / 256, 256, 0, stream>>>(Wr, wrt, HH * SS);
  k_cvt_t<<<(HH * 64 + 255) / 256, 256, 0, stream>>>(Wh, wht, HH);
  k_cvt_t<<<(SS * 64 + 255) / 256, 256, 0, stream>>>(Ww, wwt, SS);

  k_gather<<<BB * SS, 64, 0, stream>>>(mid_his, mask, emb, pos_emb, item_his, xh, iln);
  k_hid<<<2048, 256, 0, stream>>>(xh, w1t, b1, hid);
  k_logits<<<512, 256, 0, stream>>>(hid, W2, b2, mask, logits);
  k_softmax<<<512, 256, 0, stream>>>(logits, pw, att);
  k_wi<<<512, 64, 0, stream>>>(pw, item_his, wi, wih, wn);
  k_attend<<<512, 256, 0, stream>>>(wi, item_his, mask, att, pnum, pden);
  k_contrast<<<512, 256, 0, stream>>>(wn, iln, pw, mask, mid_his, pwi, pcon);
  k_recons<<<4096, 256, 0, stream>>>(wih, wrt, br, reconsh, reconsT);
  k_rh<<<4096, 256, 0, stream>>>(reconsh, wht, bh, rhh);
  k_sstats<<<1024, 256, 0, stream>>>(rhh, wwt, bw, mid_his, stats);
  k_sout<<<1024, 256, 0, stream>>>(rhh, wwt, bw, mid_his, stats, reconsT,
                                   item_his, pwi, mask, pcons);
  k_readout<<<BB, 64, 0, stream>>>(wi, emb, mid_batch, out);
  k_finalize<<<1, 256, 0, stream>>>(pnum, pden, pcon, pcons, out);

  (void)in_sizes; (void)n_in; (void)out_size; (void)ws_size;
}
